// MultiQueryAttention_3332894622598
// MI455X (gfx1250) — matrix operand order repaired and checked by reading, not compiled, not run on hardware
//
#include <hip/hip_runtime.h>
#include <hip/hip_bf16.h>
#include <math.h>

// ---------------------------------------------------------------------------
// MultiQueryAttention for MI455X (gfx1250), wave32 + WMMA f16 (f32 accum).
// B=2, S=2048, EMBED=1024, H=16, HEAD_DIM=VALUE_DIM=64.
// ---------------------------------------------------------------------------

typedef _Float16 half_t;
typedef __attribute__((ext_vector_type(16))) _Float16 v16h;
typedef __attribute__((ext_vector_type(8)))  float    v8f;

#define BATCH     2
#define SEQ       2048
#define EMBED     1024
#define NHEADS    16
#define HDIM      64
#define MROWS     (BATCH * SEQ)   // 4096
#define MASKFILL  (-10000.0f)

__device__ __forceinline__ v8f wmma_f16(v16h a, v16h b, v8f c) {
    // v_wmma_f32_16x16x32_f16  D = A(16x32) * B(32x16) + C(16x16 f32)
    return __builtin_amdgcn_wmma_f32_16x16x32_f16(
        /*neg_a=*/false, a, /*neg_b=*/false, b,
        /*c_mod=*/(short)0, c, /*reuse_a=*/false, /*reuse_b=*/false);
}

__device__ __forceinline__ v8f v8f_zero() {
    v8f z;
#pragma unroll
    for (int i = 0; i < 8; ++i) z[i] = 0.0f;
    return z;
}

// A-fragment (16x32 f16) per-lane K index: lanes 0-15 hold K {0..15},
// lanes 16-31 hold K {16..31} (contiguous order, matching the B fragments).
__device__ __forceinline__ int a_klocal(int g, int j) {
    return g * 16 + j;
}

// ---------------------------------------------------------------------------
// GEMM: C_f32[M,N] = A[M,K] * B_f32[K,N] (+bias).  A is f32 or f16 (two
// kernels below share this body via macro).  Block tile 128x128, 8 waves of
// 32x64.  K chunk = 64 (two WMMA K-steps per LDS round-trip).  B tile stored
// TRANSPOSED in LDS so B-fragments are per-lane contiguous (ds_load_b128).
// ---------------------------------------------------------------------------
#define BM 128
#define BN 128
#define BKC 64
#define LPAD 8

#define GEMM_BODY(LOAD_A_ELT, STORE_C)                                          \
    __shared__ half_t As[BM][BKC + LPAD];          /* row-major  [m][k] */      \
    __shared__ half_t BsT[BN][BKC + LPAD];         /* transposed [n][k] */      \
    const int tid  = threadIdx.x;                                               \
    const int lane = tid & 31;                                                  \
    const int wid  = tid >> 5;                                                  \
    const int g    = lane >> 4;                                                 \
    const int ln   = lane & 15;                                                 \
    const int m0   = blockIdx.x * BM;                                           \
    const int n0   = blockIdx.y * BN;                                           \
    const int wm   = (wid & 3) * 32;                                            \
    const int wn   = (wid >> 2) * 64;                                           \
    v8f acc[2][4];                                                              \
    _Pragma("unroll")                                                           \
    for (int mi = 0; mi < 2; ++mi)                                              \
        _Pragma("unroll")                                                       \
        for (int ni = 0; ni < 4; ++ni) acc[mi][ni] = v8f_zero();                \
    for (int k0 = 0; k0 < K; k0 += BKC) {                                       \
        /* prefetch next chunk (global_prefetch_b8) */                          \
        if (k0 + BKC < K) {                                                     \
            __builtin_prefetch(A  + (size_t)(m0 + (tid >> 1)) * K + k0 + BKC, 0, 1); \
            __builtin_prefetch(Bw + (size_t)(k0 + BKC + (tid >> 3)) * N + n0, 0, 1); \
        }                                                                       \
        /* stage A 128x64: thread row=tid/2, 32 contiguous elements */          \
        {                                                                       \
            const int ar = tid >> 1;                                            \
            const int ac = (tid & 1) * 32;                                      \
            _Pragma("unroll")                                                   \
            for (int j = 0; j < 32; ++j)                                        \
                As[ar][ac + j] = LOAD_A_ELT(A[(size_t)(m0 + ar) * K + k0 + ac + j]); \
        }                                                                       \
        /* stage B 64x128 -> BsT[n][k]: thread col=tid&127, 32 k-values.  */    \
        /* Global reads coalesced across threads (fixed k, consecutive n). */   \
        {                                                                       \
            const int col  = tid & 127;                                         \
            const int koff = (tid >> 7) * 32;                                   \
            const bool inb = (n0 + col < N);                                    \
            _Pragma("unroll")                                                   \
            for (int j = 0; j < 32; ++j)                                        \
                BsT[col][koff + j] =                                            \
                    inb ? (half_t)Bw[(size_t)(k0 + koff + j) * N + n0 + col]    \
                        : (half_t)0.0f;                                         \
        }                                                                       \
        __syncthreads();                                                        \
        _Pragma("unroll")                                                       \
        for (int ks = 0; ks < 2; ++ks) {                                        \
            v16h afr[2], bfr[4];                                                \
            _Pragma("unroll")                                                   \
            for (int mi = 0; mi < 2; ++mi) {                                    \
                const int row = wm + mi * 16 + ln;                              \
                _Pragma("unroll")                                               \
                for (int j = 0; j < 16; ++j)                                    \
                    afr[mi][j] = As[row][ks * 32 + a_klocal(g, j)];             \
            }                                                                   \
            _Pragma("unroll")                                                   \
            for (int ni = 0; ni < 4; ++ni) {                                    \
                const int col = wn + ni * 16 + ln;                              \
                _Pragma("unroll")                                               \
                for (int j = 0; j < 16; ++j)                                    \
                    bfr[ni][j] = BsT[col][ks * 32 + g * 16 + j];                \
            }                                                                   \
            _Pragma("unroll")                                                   \
            for (int mi = 0; mi < 2; ++mi)                                      \
                _Pragma("unroll")                                               \
                for (int ni = 0; ni < 4; ++ni)                                  \
                    acc[mi][ni] = wmma_f16(afr[mi], bfr[ni], acc[mi][ni]);      \
        }                                                                       \
        __syncthreads();                                                        \
    }                                                                           \
    _Pragma("unroll")                                                           \
    for (int mi = 0; mi < 2; ++mi)                                              \
        _Pragma("unroll")                                                       \
        for (int ni = 0; ni < 4; ++ni)                                          \
            _Pragma("unroll")                                                   \
            for (int r = 0; r < 8; ++r) {                                       \
                const int row = m0 + wm + mi * 16 + g * 8 + r;                  \
                const int col = n0 + wn + ni * 16 + ln;                         \
                if (col < N) STORE_C;                                           \
            }

#define LOADA_F32(e) ((half_t)(e))
#define LOADA_F16(e) (e)

__global__ __launch_bounds__(256) void gemm_f32f32(const float* __restrict__ A,
                                                   const float* __restrict__ Bw,
                                                   float* __restrict__ C,
                                                   int M, int N, int K) {
    GEMM_BODY(LOADA_F32, C[(size_t)row * N + col] = acc[mi][ni][r])
}

__global__ __launch_bounds__(256) void gemm_f16f32_bias(const half_t* __restrict__ A,
                                                        const float* __restrict__ Bw,
                                                        const float* __restrict__ bias,
                                                        float* __restrict__ C,
                                                        int M, int N, int K) {
    GEMM_BODY(LOADA_F16, C[(size_t)row * N + col] = acc[mi][ni][r] + bias[col])
}

// ---------------------------------------------------------------------------
// RoPE (+ f32->f16 cvt) for q,k.  src/dst are [B*S*heads*64].
// rotate pairs (d, d+32): out_d = x1*cos - x2*sin ; out_{d+32} = x1*sin + x2*cos
// ---------------------------------------------------------------------------
__global__ void rope_cvt(const float* __restrict__ src, half_t* __restrict__ dst,
                         long total, int heads) {
    long i = (long)blockIdx.x * blockDim.x + threadIdx.x;
    if (i >= total) return;
    const int d = (int)(i & 63);
    const long row = i >> 6;                       // (b*S + s)*heads + h
    const int s = (int)((row / heads) % SEQ);      // sequence position
    float out;
    if (d < 32) {
        const float theta = (float)s / __powf(10000.0f, (float)d / 32.0f);
        float sn, cs; __sincosf(theta, &sn, &cs);
        out = src[i] * cs - src[i + 32] * sn;
    } else {
        const float theta = (float)s / __powf(10000.0f, (float)(d - 32) / 32.0f);
        float sn, cs; __sincosf(theta, &sn, &cs);
        out = src[i - 32] * sn + src[i] * cs;
    }
    dst[i] = (half_t)out;
}

// V: f32 [B,S,64] -> f16 TRANSPOSED [B,64,S] so PV B-fragments are per-lane
// contiguous along the key dimension (global_load_b128 in flash kernel).
__global__ void cvt_v_transpose(const float* __restrict__ src, half_t* __restrict__ dst) {
    long o = (long)blockIdx.x * blockDim.x + threadIdx.x;  // over B*64*S
    if (o >= (long)BATCH * HDIM * SEQ) return;
    const int s = (int)(o % SEQ);
    const int d = (int)((o / SEQ) % HDIM);
    const int b = (int)(o / ((long)SEQ * HDIM));
    dst[o] = (half_t)src[((size_t)(b * SEQ + s)) * HDIM + d];
}

// ---------------------------------------------------------------------------
// Flash attention.  Block = 128 threads = 4 waves; each wave owns a 16-row Q
// tile of one (b,h).  Streams keys in chunks of 32 with online softmax.
//   logits = tanh((q.k/8 + bias)/5)*5, masked, softmax, O = P @ V.
// ---------------------------------------------------------------------------
__global__ __launch_bounds__(128) void flash_attn(const half_t* __restrict__ qh,
                                                  const half_t* __restrict__ kh,
                                                  const half_t* __restrict__ vT,
                                                  const float* __restrict__ bias,
                                                  const unsigned char* __restrict__ mask,
                                                  half_t* __restrict__ attn) {
    __shared__ half_t Pt[4][16][32 + LPAD];   // per-wave P tile (C->A layout bounce)
    const int tid  = threadIdx.x;
    const int lane = tid & 31;
    const int wv   = tid >> 5;
    const int g    = lane >> 4;
    const int ln   = lane & 15;

    const int blocks_per_bh = SEQ / 64;                     // 32
    const int bh = blockIdx.x / blocks_per_bh;
    const int qb = blockIdx.x % blocks_per_bh;
    const int b  = bh / NHEADS;
    const int h  = bh % NHEADS;
    const int s0 = qb * 64 + wv * 16;

    // Q fragments (two K=32 steps across HEAD_DIM=64), A-layout; contiguous reads.
    v16h aq0, aq1;
    {
        const half_t* qrow = qh + ((size_t)(b * SEQ + s0 + ln) * NHEADS + h) * HDIM;
#pragma unroll
        for (int j = 0; j < 16; ++j) {
            const int kl = a_klocal(g, j);
            aq0[j] = qrow[kl];
            aq1[j] = qrow[32 + kl];
        }
    }

    // Query-row mask (loop invariant): MASKFILL whole row when query masked.
    bool qmask[8];
    const float* biasrow[8];
#pragma unroll
    for (int r = 0; r < 8; ++r) {
        const int srow = s0 + g * 8 + r;
        qmask[r]   = (mask[b * SEQ + srow] != 0);
        biasrow[r] = bias + ((size_t)b * SEQ + srow) * SEQ;
    }

    float mrow[8], lrow[8];
    v8f oacc[4];
#pragma unroll
    for (int r = 0; r < 8; ++r) { mrow[r] = -1e30f; lrow[r] = 0.0f; }
#pragma unroll
    for (int sub = 0; sub < 4; ++sub) oacc[sub] = v8f_zero();

    for (int t0 = 0; t0 < SEQ; t0 += 32) {
        // Key-padding for this chunk's two columns handled by this lane.
        const int ta = t0 + ln, tb = t0 + 16 + ln;
        const bool kma = (mask[b * SEQ + ta] != 0);
        const bool kmb = (mask[b * SEQ + tb] != 0);

        // K^T fragments (B-layout): lane column = key; contiguous 32B reads.
        v16h bk00, bk01, bk10, bk11;
        {
            const half_t* kr0 = kh + (size_t)(b * SEQ + ta) * HDIM;
            const half_t* kr1 = kh + (size_t)(b * SEQ + tb) * HDIM;
#pragma unroll
            for (int j = 0; j < 16; ++j) {
                const int kd = g * 16 + j;
                bk00[j] = kr0[kd];      bk01[j] = kr0[32 + kd];
                bk10[j] = kr1[kd];      bk11[j] = kr1[32 + kd];
            }
        }
        v8f sa0 = v8f_zero(), sa1 = v8f_zero();
        sa0 = wmma_f16(aq0, bk00, sa0);  sa0 = wmma_f16(aq1, bk01, sa0);
        sa1 = wmma_f16(aq0, bk10, sa1);  sa1 = wmma_f16(aq1, bk11, sa1);

        // Scale + bias + softcap + mask (C-layout: row = g*8+r, col = ln).
        float p0[8], p1[8];
#pragma unroll
        for (int r = 0; r < 8; ++r) {
            float l0 = sa0[r] * 0.125f + biasrow[r][ta];
            float l1 = sa1[r] * 0.125f + biasrow[r][tb];
            l0 = 5.0f * tanhf(l0 * 0.2f);
            l1 = 5.0f * tanhf(l1 * 0.2f);
            if (!qmask[r] || !kma) l0 = MASKFILL;
            if (!qmask[r] || !kmb) l1 = MASKFILL;
            p0[r] = l0; p1[r] = l1;
        }
        // Online softmax: per-row reduce across the 16 lanes of each half.
#pragma unroll
        for (int r = 0; r < 8; ++r) {
            float mx = fmaxf(p0[r], p1[r]);
#pragma unroll
            for (int d = 1; d < 16; d <<= 1) mx = fmaxf(mx, __shfl_xor(mx, d));
            const float mnew  = fmaxf(mrow[r], mx);
            const float alpha = __expf(mrow[r] - mnew);
            const float e0 = __expf(p0[r] - mnew);
            const float e1 = __expf(p1[r] - mnew);
            float sum = e0 + e1;
#pragma unroll
            for (int d = 1; d < 16; d <<= 1) sum += __shfl_xor(sum, d);
            lrow[r] = lrow[r] * alpha + sum;
            mrow[r] = mnew;
#pragma unroll
            for (int sub = 0; sub < 4; ++sub) oacc[sub][r] *= alpha;
            Pt[wv][g * 8 + r][ln]      = (half_t)e0;
            Pt[wv][g * 8 + r][16 + ln] = (half_t)e1;
        }
        __syncthreads();
        // Reload P in A-layout (contiguous 8+8 halfs -> ds_load_b128 x2).
        v16h ap;
#pragma unroll
        for (int j = 0; j < 16; ++j) ap[j] = Pt[wv][ln][a_klocal(g, j)];

        // PV with transposed V: per-lane contiguous along keys.
#pragma unroll
        for (int sub = 0; sub < 4; ++sub) {
            const half_t* vrow = vT + ((size_t)b * HDIM + sub * 16 + ln) * SEQ + t0;
            v16h bv;
#pragma unroll
            for (int j = 0; j < 16; ++j) bv[j] = vrow[g * 16 + j];
            oacc[sub] = wmma_f16(ap, bv, oacc[sub]);
        }
        __syncthreads();
    }

    // Epilogue: normalize and store f16 attn output [B,S,H*64].
#pragma unroll
    for (int sub = 0; sub < 4; ++sub)
#pragma unroll
        for (int r = 0; r < 8; ++r) {
            const int srow = s0 + g * 8 + r;
            const float v = oacc[sub][r] / lrow[r];
            attn[(size_t)(b * SEQ + srow) * (NHEADS * HDIM) + h * HDIM + sub * 16 + ln] =
                (half_t)v;
        }
}

// ---------------------------------------------------------------------------
// Launch
// ---------------------------------------------------------------------------
extern "C" void kernel_launch(void* const* d_in, const int* in_sizes, int n_in,
                              void* d_out, int out_size, void* d_ws, size_t ws_size,
                              hipStream_t stream) {
    (void)in_sizes; (void)n_in; (void)out_size; (void)ws_size;
    const float*         x     = (const float*)d_in[0];
    const float*         bias  = (const float*)d_in[1];
    const unsigned char* kmask = (const unsigned char*)d_in[2];  // bool array (1B/elt)
    const float*         Wq    = (const float*)d_in[3];
    const float*         Wk    = (const float*)d_in[4];
    const float*         Wv    = (const float*)d_in[5];
    const float*         Wout  = (const float*)d_in[6];
    const float*         b_out = (const float*)d_in[7];
    float*               out   = (float*)d_out;

    // Workspace carve-up (~37 MB).
    char* w = (char*)d_ws;
    float*  q32 = (float*)w;   w += (size_t)MROWS * EMBED * sizeof(float);   // 16 MB
    float*  k32 = (float*)w;   w += (size_t)MROWS * HDIM  * sizeof(float);   // 1 MB
    float*  v32 = (float*)w;   w += (size_t)MROWS * HDIM  * sizeof(float);   // 1 MB
    half_t* qhb = (half_t*)w;  w += (size_t)MROWS * EMBED * sizeof(half_t);  // 8 MB
    half_t* khb = (half_t*)w;  w += (size_t)MROWS * HDIM  * sizeof(half_t);
    half_t* vTb = (half_t*)w;  w += (size_t)MROWS * HDIM  * sizeof(half_t);  // [B,64,S]
    half_t* ahb = (half_t*)w;  w += (size_t)MROWS * EMBED * sizeof(half_t);  // 8 MB

    // 1) Projections (WMMA GEMMs, f32 out).
    gemm_f32f32<<<dim3(MROWS / BM, EMBED / BN), 256, 0, stream>>>(x, Wq, q32, MROWS, EMBED, EMBED);
    gemm_f32f32<<<dim3(MROWS / BM, 1),          256, 0, stream>>>(x, Wk, k32, MROWS, HDIM, EMBED);
    gemm_f32f32<<<dim3(MROWS / BM, 1),          256, 0, stream>>>(x, Wv, v32, MROWS, HDIM, EMBED);

    // 2) RoPE + f16 conversion (V transposed for flash PV fragments).
    {
        const long nq = (long)MROWS * EMBED;
        const long nk = (long)MROWS * HDIM;
        rope_cvt<<<(unsigned)((nq + 255) / 256), 256, 0, stream>>>(q32, qhb, nq, NHEADS);
        rope_cvt<<<(unsigned)((nk + 255) / 256), 256, 0, stream>>>(k32, khb, nk, 1);
        cvt_v_transpose<<<(unsigned)((nk + 255) / 256), 256, 0, stream>>>(v32, vTb);
    }

    // 3) Flash attention: B*H*(S/64) blocks of 4 waves.
    flash_attn<<<BATCH * NHEADS * (SEQ / 64), 128, 0, stream>>>(qhb, khb, vTb, bias, kmask, ahb);

    // 4) Output projection + bias (WMMA GEMM, f32 out).
    gemm_f16f32_bias<<<dim3(MROWS / BM, EMBED / BN), 256, 0, stream>>>(ahb, Wout, b_out, out,
                                                                      MROWS, EMBED, EMBED);
}
